// MultiHeadAttentionSimple_31430570672459
// MI455X (gfx1250) — matrix operand order repaired and checked by reading, not compiled, not run on hardware
//
#include <hip/hip_runtime.h>
#include <hip/hip_bf16.h>

typedef _Float16 h16;
typedef _Float16 v4h  __attribute__((ext_vector_type(4)));
typedef _Float16 v8h  __attribute__((ext_vector_type(8)));
typedef _Float16 v16h __attribute__((ext_vector_type(16)));
typedef float    v8f  __attribute__((ext_vector_type(8)));

#define D_MODEL 1024
#define NHEAD   16
#define D_HEAD  64
#define SEQ     1024
#define BATCH   4
#define MROWS   (BATCH * SEQ)   // 4096

// ---------------------------------------------------------------------------
// helpers
// ---------------------------------------------------------------------------
__device__ __forceinline__ v16h cat16(v8h lo, v8h hi) {
    v16h r;
#pragma unroll
    for (int i = 0; i < 8; ++i) { r[i] = lo[i]; r[i + 8] = hi[i]; }
    return r;
}

// 16 contiguous halves -> v16h (two b128 loads)
__device__ __forceinline__ v16h ld16_contig(const h16* p) {
    v8h a = *(const v8h*)(p);
    v8h b = *(const v8h*)(p + 8);
    return cat16(a, b);
}

__device__ __forceinline__ v8f wmma_f16(v16h a, v16h b, v8f c) {
    return __builtin_amdgcn_wmma_f32_16x16x32_f16(
        /*neg_a=*/false, a, /*neg_b=*/false, b,
        /*c_mod=*/(short)0, c, /*reuse_a=*/false, /*reuse_b=*/false);
}

// ---------------------------------------------------------------------------
// prep: f32 -> f16 copy (coalesced both ways)
// ---------------------------------------------------------------------------
__global__ __launch_bounds__(256) void cvt_f32_f16(const float* __restrict__ src,
                                                   h16* __restrict__ dst, int n) {
    int i = (blockIdx.x * 256 + threadIdx.x) * 4;
    if (i + 3 < n) {
        float4 f = *(const float4*)(src + i);
        v4h o = { (h16)f.x, (h16)f.y, (h16)f.z, (h16)f.w };
        *(v4h*)(dst + i) = o;
    }
}

// ---------------------------------------------------------------------------
// prep: f32 [K,N] -> f16 [N,K] transpose, 64x64 LDS tiles.
// Coalesced f32 reads, coalesced f16 writes; LDS row stride 66 -> no conflicts.
// ---------------------------------------------------------------------------
__global__ __launch_bounds__(256) void transpose_f32_f16(const float* __restrict__ src,
                                                         h16* __restrict__ dst,
                                                         int K, int N) {
    __shared__ h16 tile[64][66];
    const int n0 = blockIdx.x * 64;
    const int k0 = blockIdx.y * 64;
    const int tx = threadIdx.x & 63;   // fast axis
    const int ty = threadIdx.x >> 6;   // 0..3
#pragma unroll
    for (int i = 0; i < 64; i += 4)
        tile[ty + i][tx] = (h16)src[(size_t)(k0 + ty + i) * N + n0 + tx];
    __syncthreads();
#pragma unroll
    for (int i = 0; i < 64; i += 4)
        dst[(size_t)(n0 + ty + i) * K + k0 + tx] = tile[tx][ty + i];
}

// ---------------------------------------------------------------------------
// WMMA GEMM:  out = A[M,K](f16 row-major) @ Bt[N,K](f16 row-major = B^T) + bias
// wave tile 32x64 (2 M-tiles x 4 N-tiles), 8 waves/block -> 256x64 per block,
// grid = (M/256, N/64). Manual ping-pong double buffer (unroll-by-2): no
// register rotation copies; prefetch of chunk i+1 overlaps WMMAs of chunk i.
// mode 0: f16 row-major   mode 1: f16 per-head transposed [B,H,Dh,S] (v8h store)
// mode 2: f32 row-major
// ---------------------------------------------------------------------------
__global__ __launch_bounds__(256) void gemm_wmma(const h16* __restrict__ A,
                                                 const h16* __restrict__ Bt,
                                                 const float* __restrict__ bias,
                                                 void* __restrict__ out,
                                                 int M, int N, int K,
                                                 int mode, float scale) {
    const int lane = threadIdx.x & 31;
    const int w    = threadIdx.x >> 5;
    const int hi   = lane >> 4;
    const int ln   = lane & 15;

    const int m0 = blockIdx.x * 256 + w * 32;
    const int n0 = blockIdx.y * 64;

    v8f zero = {};
    v8f acc[2][4];
#pragma unroll
    for (int mt = 0; mt < 2; ++mt)
#pragma unroll
        for (int t = 0; t < 4; ++t) acc[mt][t] = zero;

    const h16* ar0 = A + (size_t)(m0 + ln) * K;
    const h16* ar1 = A + (size_t)(m0 + 16 + ln) * K;
    const h16* bc  = Bt + (size_t)(n0 + ln) * K + hi * 16;

    v16h a0[2], a1[2], bb[2][4];

    auto ldTiles = [&](int buf, int kk) {
        a0[buf] = ld16_contig(ar0 + kk + hi * 16);
        a1[buf] = ld16_contig(ar1 + kk + hi * 16);
#pragma unroll
        for (int t = 0; t < 4; ++t)
            bb[buf][t] = ld16_contig(bc + (size_t)t * 16 * K + kk);
    };
    auto compute = [&](int buf) {
#pragma unroll
        for (int t = 0; t < 4; ++t) {
            acc[0][t] = wmma_f16(a0[buf], bb[buf][t], acc[0][t]);
            acc[1][t] = wmma_f16(a1[buf], bb[buf][t], acc[1][t]);
        }
    };

    ldTiles(0, 0);
#pragma unroll 1
    for (int kk = 0; kk < K - 64; kk += 64) {
        ldTiles(1, kk + 32);
        compute(0);
        ldTiles(0, kk + 64);
        compute(1);
    }
    ldTiles(1, K - 32);
    compute(0);
    compute(1);

    // epilogue: C layout -> row = m0 + mt*16 + v + 8*hi, col = n0 + t*16 + ln
    if (mode == 1) {
        // r = b*SEQ + s ; c = h*64 + d ; Vt[b][h][d][s]; v=0..7 -> 8 consecutive s
#pragma unroll
        for (int mt = 0; mt < 2; ++mt) {
#pragma unroll
            for (int t = 0; t < 4; ++t) {
                int c  = n0 + t * 16 + ln;
                float bc2 = bias[c];
                int r0 = m0 + mt * 16 + hi * 8;          // first of 8 consecutive rows
                int bb2 = r0 >> 10, s0 = r0 & 1023;
                int hh  = c >> 6,   d  = c & 63;
                v8h pack;
#pragma unroll
                for (int v = 0; v < 8; ++v)
                    pack[v] = (h16)((acc[mt][t][v] + bc2) * scale);
                *(v8h*)(&((h16*)out)[((size_t)(((bb2 << 4) + hh) << 6) + d) * SEQ + s0]) = pack;
            }
        }
    } else {
#pragma unroll
        for (int mt = 0; mt < 2; ++mt) {
#pragma unroll
            for (int t = 0; t < 4; ++t) {
                int c = n0 + t * 16 + ln;
                float bc2 = bias[c];
#pragma unroll
                for (int v = 0; v < 8; ++v) {
                    int r = m0 + mt * 16 + v + hi * 8;
                    float val = (acc[mt][t][v] + bc2) * scale;
                    if (mode == 0) ((h16*)out)[(size_t)r * N + c] = (h16)val;
                    else           ((float*)out)[(size_t)r * N + c] = val;
                }
            }
        }
    }
}

// ---------------------------------------------------------------------------
// Flash attention: one wave = 16 query rows of one (b,h).
// Q pre-scaled by log2(e)/sqrt(Dh) so softmax uses exp2.
// Qh,Kh: f16 [B*S, D_MODEL]; Vt: f16 [B,H,Dh,S]; Ctx: f16 [B*S, D_MODEL]
// K tiles ping-pong between two register buffers (unroll-by-2, no copies);
// V loads and next-K prefetch are issued before the softmax VALU section.
// ---------------------------------------------------------------------------
__global__ __launch_bounds__(256) void attn_wmma(const h16* __restrict__ Qh,
                                                 const h16* __restrict__ Kh,
                                                 const h16* __restrict__ Vt,
                                                 h16* __restrict__ Ctx) {
    __shared__ h16 lds[8 * 16 * 32];          // per-wave 16x32 P tile
    const int lane = threadIdx.x & 31;
    const int w    = threadIdx.x >> 5;
    const int hi   = lane >> 4;
    const int ln   = lane & 15;

    const int g   = blockIdx.x * 8 + w;       // 0..4095
    const int b   = g >> 10;
    const int rem = g & 1023;
    const int h   = rem >> 6;
    const int q0  = (rem & 63) << 4;

    h16* plds = lds + w * (16 * 32);

    const h16* qbase = Qh + (size_t)((b << 10) + q0 + ln) * D_MODEL + (h << 6);
    v16h aq0 = ld16_contig(qbase + hi * 16);
    v16h aq1 = ld16_contig(qbase + 32 + hi * 16);

    v8f zero = {};
    v8f acc[4];
#pragma unroll
    for (int t = 0; t < 4; ++t) acc[t] = zero;
    float mrow[8], lrow[8];
#pragma unroll
    for (int v = 0; v < 8; ++v) { mrow[v] = -3.0e38f; lrow[v] = 0.0f; }

    const h16* kbase = Kh + (size_t)(b << 10) * D_MODEL + (h << 6);
    const h16* vbase = Vt + (size_t)(((b << 4) + h) << 6) * SEQ;

    v16h bufK[2][4];
    auto ldK = [&](int buf, int KB) {
        const h16* c0 = kbase + (size_t)(KB + ln) * D_MODEL;
        const h16* c1 = kbase + (size_t)(KB + 16 + ln) * D_MODEL;
        bufK[buf][0] = ld16_contig(c0 + hi * 16);
        bufK[buf][1] = ld16_contig(c0 + 32 + hi * 16);
        bufK[buf][2] = ld16_contig(c1 + hi * 16);
        bufK[buf][3] = ld16_contig(c1 + 32 + hi * 16);
    };

    auto step = [&](int cur, int nxt, int kb) {
        // ---- scores for 32 keys -------------------------------------------
        v8f sc0 = zero, sc1 = zero;
        sc0 = wmma_f16(aq0, bufK[cur][0], sc0);
        sc0 = wmma_f16(aq1, bufK[cur][1], sc0);
        sc1 = wmma_f16(aq0, bufK[cur][2], sc1);
        sc1 = wmma_f16(aq1, bufK[cur][3], sc1);

        // ---- issue V loads + next-K prefetch before the VALU softmax ------
        v16h bv[4];
#pragma unroll
        for (int t = 0; t < 4; ++t)
            bv[t] = ld16_contig(vbase + (size_t)(t * 16 + ln) * SEQ + kb + hi * 16);
        ldK(nxt, (kb + 32) & (SEQ - 1));   // wrapped: final prefetch harmless

        // ---- online softmax (row = v + 8*hi, key = ln / 16+ln) ------------
#pragma unroll
        for (int v = 0; v < 8; ++v) {
            float s0 = sc0[v], s1 = sc1[v];
            float bm = fmaxf(s0, s1);
#pragma unroll
            for (int msk = 1; msk < 16; msk <<= 1)
                bm = fmaxf(bm, __shfl_xor(bm, msk, 32));
            float mnew = fmaxf(mrow[v], bm);
            float corr = exp2f(mrow[v] - mnew);
            mrow[v] = mnew;
            float p0 = exp2f(s0 - mnew);
            float p1 = exp2f(s1 - mnew);
            float ps = p0 + p1;
#pragma unroll
            for (int msk = 1; msk < 16; msk <<= 1)
                ps += __shfl_xor(ps, msk, 32);
            lrow[v] = lrow[v] * corr + ps;
#pragma unroll
            for (int t = 0; t < 4; ++t) acc[t][v] *= corr;
            int r = v + hi * 8;
            plds[r * 32 + ln]      = (h16)p0;
            plds[r * 32 + 16 + ln] = (h16)p1;
        }
        __builtin_amdgcn_wave_barrier();   // same-wave DS ops are in-order

        // ---- P (A-layout) from LDS, then P @ V ----------------------------
        v16h ap = ld16_contig(plds + ln * 32 + hi * 16);
#pragma unroll
        for (int t = 0; t < 4; ++t)
            acc[t] = wmma_f16(ap, bv[t], acc[t]);
        __builtin_amdgcn_wave_barrier();
    };

    ldK(0, 0);
#pragma unroll 1
    for (int kb = 0; kb < SEQ; kb += 64) {
        step(0, 1, kb);
        step(1, 0, kb + 32);
    }

    // ---- normalize + store context (f16) ----------------------------------
    float inv[8];
#pragma unroll
    for (int v = 0; v < 8; ++v) inv[v] = 1.0f / lrow[v];
#pragma unroll
    for (int t = 0; t < 4; ++t) {
#pragma unroll
        for (int v = 0; v < 8; ++v) {
            int r = q0 + v + hi * 8;
            Ctx[(size_t)((b << 10) + r) * D_MODEL + (h << 6) + t * 16 + ln] =
                (h16)(acc[t][v] * inv[v]);
        }
    }
}

// ---------------------------------------------------------------------------
// host launcher
// ---------------------------------------------------------------------------
extern "C" void kernel_launch(void* const* d_in, const int* in_sizes, int n_in,
                              void* d_out, int out_size, void* d_ws, size_t ws_size,
                              hipStream_t stream) {
    (void)in_sizes; (void)n_in; (void)out_size; (void)ws_size;

    const float* q_in = (const float*)d_in[0];
    const float* k_in = (const float*)d_in[1];
    const float* v_in = (const float*)d_in[2];
    const float* Wq   = (const float*)d_in[3];
    const float* bq   = (const float*)d_in[4];
    const float* Wk   = (const float*)d_in[5];
    const float* bk   = (const float*)d_in[6];
    const float* Wv   = (const float*)d_in[7];
    const float* bv   = (const float*)d_in[8];
    const float* Wo   = (const float*)d_in[9];
    const float* bo   = (const float*)d_in[10];

    char* ws = (char*)d_ws;
    const size_t MB = 1ull << 20;
    h16* Xq  = (h16*)(ws + 0 * MB);
    h16* Xk  = (h16*)(ws + 8 * MB);
    h16* Xv  = (h16*)(ws + 16 * MB);
    h16* WqT = (h16*)(ws + 24 * MB);
    h16* WkT = (h16*)(ws + 26 * MB);
    h16* WvT = (h16*)(ws + 28 * MB);
    h16* WoT = (h16*)(ws + 30 * MB);
    h16* Qh  = (h16*)(ws + 32 * MB);   // pre-scaled by log2(e)/8
    h16* Khb = (h16*)(ws + 40 * MB);
    h16* Vt  = (h16*)(ws + 48 * MB);   // [B,H,Dh,S]
    h16* Ctx = (h16*)(ws + 56 * MB);

    const int NACT = MROWS * D_MODEL;          // 4194304

    cvt_f32_f16<<<NACT / (256 * 4), 256, 0, stream>>>(q_in, Xq, NACT);
    cvt_f32_f16<<<NACT / (256 * 4), 256, 0, stream>>>(k_in, Xk, NACT);
    cvt_f32_f16<<<NACT / (256 * 4), 256, 0, stream>>>(v_in, Xv, NACT);

    dim3 tgrid(D_MODEL / 64, D_MODEL / 64);
    transpose_f32_f16<<<tgrid, 256, 0, stream>>>(Wq, WqT, D_MODEL, D_MODEL);
    transpose_f32_f16<<<tgrid, 256, 0, stream>>>(Wk, WkT, D_MODEL, D_MODEL);
    transpose_f32_f16<<<tgrid, 256, 0, stream>>>(Wv, WvT, D_MODEL, D_MODEL);
    transpose_f32_f16<<<tgrid, 256, 0, stream>>>(Wo, WoT, D_MODEL, D_MODEL);

    dim3 ggrid(MROWS / 256, D_MODEL / 64);     // (16,16)
    const float qscale = 1.4426950408889634f / 8.0f;   // log2(e)/sqrt(Dh)
    gemm_wmma<<<ggrid, 256, 0, stream>>>(Xq, WqT, bq, Qh,  MROWS, D_MODEL, D_MODEL, 0, qscale);
    gemm_wmma<<<ggrid, 256, 0, stream>>>(Xk, WkT, bk, Khb, MROWS, D_MODEL, D_MODEL, 0, 1.0f);
    gemm_wmma<<<ggrid, 256, 0, stream>>>(Xv, WvT, bv, Vt,  MROWS, D_MODEL, D_MODEL, 1, 1.0f);

    attn_wmma<<<(BATCH * NHEAD * (SEQ / 16)) / 8, 256, 0, stream>>>(Qh, Khb, Vt, Ctx);

    gemm_wmma<<<ggrid, 256, 0, stream>>>(Ctx, WoT, bo, d_out, MROWS, D_MODEL, D_MODEL, 2, 1.0f);
}
